// TConvDRAW_8246337208467
// MI455X (gfx1250) — matrix operand order repaired and checked by reading, not compiled, not run on hardware
//
#include <hip/hip_runtime.h>
#include <stdint.h>

// ---------------- types ----------------
typedef __attribute__((ext_vector_type(16))) __bf16   v16bf;
typedef __attribute__((ext_vector_type(8)))  float    v8f;
typedef __attribute__((ext_vector_type(4)))  uint32_t v4u;
typedef __attribute__((ext_vector_type(8)))  int      v8i;
typedef __attribute__((ext_vector_type(4)))  int      v4i;

union ABfrag { v16bf v; uint32_t u[8]; };

static __device__ __forceinline__ uint16_t f32_to_bf16(float f) {
  union { float f; uint32_t u; } c; c.f = f;
  uint32_t u = c.u;
  uint32_t r = (u + 0x7FFFu + ((u >> 16) & 1u)) >> 16;   // RNE
  return (uint16_t)r;
}
static __device__ __forceinline__ float sigm(float x) { return 1.0f / (1.0f + expf(-x)); }

// ---------------- constants ----------------
#define BN   128     // batch
#define HW   256     // 16x16 spatial
#define NZ   64
#define NCL  128
#define NCC  128
#define NCS  128
#define NA   8
#define STEPS 6

// LDS geometry (dynamic shared):
//  A tile: 20x20 halo x 32ci, row stride padded 32->36 halves  (conflict-free ds reads)
//  B tile: 25tap x 64co x 32ci, row stride padded 32->36 halves (matches TDM pad 2dw/16dw)
#define A_STRIDE_H 36
#define A_ELEMS    (400 * A_STRIDE_H)          // 14400 halves = 28.8 KB
#define B_ROWS     (25 * 64)
#define B_STRIDE_H 36
#define B_ELEMS    (B_ROWS * B_STRIDE_H)       // 57600 halves = 115.2 KB
#define SMEM_BYTES ((A_ELEMS + B_ELEMS) * 2)   // 144 KB  (2 blocks / 320KB WGP)

// TDM availability (device pass only; host pass falls back to manual copy -- both compile)
#if defined(__has_builtin)
#if __has_builtin(__builtin_amdgcn_tensor_load_to_lds)
#define USE_TDM 1
#endif
#endif
#ifndef USE_TDM
#define USE_TDM 0
#endif

static __device__ __forceinline__ void wait_tensorcnt0() {
#if defined(__has_builtin) && __has_builtin(__builtin_amdgcn_s_wait_tensorcnt)
  __builtin_amdgcn_s_wait_tensorcnt(0);
#elif defined(__gfx1250__)
  asm volatile("s_wait_tensorcnt 0x0" ::: "memory");
#endif
}

// ================== implicit-GEMM 5x5 SAME conv via WMMA bf16 ==================
// X  : [B][Cin_pad][16][16]  bf16 (pad channels zero)
// Wr : [25][Cout][Cin_pad]   bf16 (pad ci zero)
// Y  : [B][Cout][16][16]     f32  (= conv + bias)
// grid: (Cout/64, B), block: 256 = 8 waves. Wave w: M-tiles {2w,2w+1} x 4 N-tiles.
__global__ __launch_bounds__(256) void conv5x5_wmma(
    const uint16_t* __restrict__ X, const uint16_t* __restrict__ Wr,
    const float* __restrict__ bias, float* __restrict__ Y,
    int Cin_pad, int Cout)
{
  extern __shared__ uint16_t smem[];
  uint16_t* ldsA = smem;                       // [rc(400)][36]
  uint16_t* ldsB = smem + A_ELEMS;             // [tap*64+co][36]

  const int b       = blockIdx.y;
  const int co_base = blockIdx.x * 64;
  const int tid     = threadIdx.x;
  const int lane    = tid & 31;
  const int wv      = tid >> 5;
  const int n       = lane & 15;     // N column / A row within tile
  const int kg      = lane >> 4;     // K half-group select

  v8f acc[2][4];
#pragma unroll
  for (int i = 0; i < 2; ++i)
#pragma unroll
    for (int j = 0; j < 4; ++j)
#pragma unroll
      for (int e = 0; e < 8; ++e) acc[i][j][e] = 0.0f;

  const int nchunk = Cin_pad >> 5;
  for (int cc = 0; cc < nchunk; ++cc) {
    __syncthreads();   // previous chunk's LDS reads complete

    // ---- stage weight slice [25][64co][32ci] into LDS ----
#if USE_TDM
    if (tid < 32) {
      // one TDM per block: 3D tile x=16dw, y=64 rows, z=25 taps; 8B LDS pad per 64B row
      uint64_t gaddr = (uint64_t)(uintptr_t)(Wr +
          ((size_t)co_base * Cin_pad + (size_t)cc * 32));
      uint32_t lbase = (uint32_t)(uintptr_t)ldsB;          // addr[31:0] = LDS offset
      uint32_t dim0  = (uint32_t)(Cin_pad >> 1);           // dword units
      uint64_t str0  = (uint64_t)(Cin_pad >> 1);           // per-co row stride (dwords)
      uint64_t str1  = (uint64_t)Cout * (uint64_t)(Cin_pad >> 1); // per-tap stride

      v4u g0 = { 1u,                                       // count=1, user load
                 lbase,
                 (uint32_t)gaddr,
                 (uint32_t)((gaddr >> 32) & 0x01FFFFFFu) | (2u << 30) }; // type=2
      uint32_t q0 = (2u << 16)      // data_size = 4B
                  | (1u << 20)      // pad_enable
                  | (3u << 22)      // pad_interval: 16 dwords
                  | (1u << 25);     // pad_amount : 2 dwords
      v8i g1 = { (int)q0,
                 (int)((dim0 & 0xFFFFu) << 16),            // tensor_dim0 lo16
                 (int)((dim0 >> 16) | (64u << 16)),        // dim0 hi16 | tensor_dim1 lo16
                 (int)(16u << 16),                         // dim1 hi16(0) | tile_dim0=16
                 (int)(64u | (25u << 16)),                 // tile_dim1=64 | tile_dim2=25
                 (int)(uint32_t)str0,                      // stride0 lo32
                 (int)((uint32_t)(str0 >> 32) | (((uint32_t)str1 & 0xFFFFu) << 16)),
                 (int)(uint32_t)(str1 >> 16) };            // stride1 [47:16]
      v4i g2 = { 25, 0, 0, 0 };                            // tensor_dim2=25
      v4i g3 = { 0, 0, 0, 0 };
#if __has_include(<hip/amd_detail/amd_gfx1250_TDM.h>)
      v8i gz = { 0, 0, 0, 0, 0, 0, 0, 0 };
      __builtin_amdgcn_tensor_load_to_lds(g0, g1, g2, g3, gz, 0);   // clang-23 arity
#else
      __builtin_amdgcn_tensor_load_to_lds(g0, g1, g2, g3, 0);       // ROCm 7.2 arity
#endif
    }
#else
    {
      uint32_t* ldsB32 = (uint32_t*)ldsB;
      for (int e = tid; e < B_ROWS * 16; e += 256) {
        int j   = e & 15;
        int row = e >> 4;                 // tap*64 + co
        int tap = row >> 6, co = row & 63;
        const uint32_t* wp = (const uint32_t*)(Wr +
            (((size_t)tap * Cout + co_base + co) * Cin_pad + cc * 32));
        ldsB32[row * (B_STRIDE_H / 2) + j] = wp[j];
      }
    }
#endif

    // ---- prefetch next activation chunk (global_prefetch_b8) ----
    if (cc + 1 < nchunk) {
      const char* nx = (const char*)(X + ((size_t)b * Cin_pad + (size_t)(cc + 1) * 32) * HW);
      __builtin_prefetch(nx + (size_t)tid * 64, 0, 0);
    }

    // ---- build haloed A tile [r][c][k] (transpose + zero halo; TDM can't do this) ----
    {
      const uint16_t* Xb = X + ((size_t)b * Cin_pad + (size_t)cc * 32) * HW;
      for (int e = tid; e < 400 * 32; e += 256) {
        int rc = e % 400;                 // consecutive lanes -> consecutive pixels
        int k  = e / 400;
        int r = rc / 20, c = rc % 20;
        int h = r - 2, w2 = c - 2;
        uint16_t val = 0;
        if ((unsigned)h < 16u && (unsigned)w2 < 16u)
          val = Xb[(size_t)k * HW + h * 16 + w2];
        ldsA[rc * A_STRIDE_H + k] = val;
      }
    }

#if USE_TDM
    if (tid < 32) wait_tensorcnt0();      // TDM issued by wave 0 -> wave 0 waits
#endif
    __syncthreads();                      // A stores + B DMA visible to all waves

    const uint32_t* lds32A = (const uint32_t*)ldsA;
    const uint32_t* lds32B = (const uint32_t*)ldsB;

    for (int tap = 0; tap < 25; ++tap) {
      const int kh = tap / 5, kw = tap - kh * 5;

      // B fragments from LDS: row = tap*64 + co_local; 8 contiguous dwords per lane
      // B K order used here (contiguous, same as A): lanes0-15 K0-15, lanes16-31 K16-31, VGPR j -> K=2j,2j+1 in half
      ABfrag bf[4];
#pragma unroll
      for (int nt = 0; nt < 4; ++nt) {
        const int base = (tap * 64 + nt * 16 + n) * (B_STRIDE_H / 2) + kg * 8;
#pragma unroll
        for (int j = 0; j < 8; ++j) bf[nt].u[j] = lds32B[base + j];
      }

#pragma unroll
      for (int mi = 0; mi < 2; ++mi) {
        const int mt = wv * 2 + mi;       // output row h
        const int r  = mt + kh;
        const int c  = n + kw;            // A row m == lane&15 == out col w
        // A 16x32 (matches B K order): lane<16 K0-15, lane>=16 K16-31
        ABfrag af;
        const int base = (r * 20 + c) * (A_STRIDE_H / 2);
        const int kbA  = kg * 8;
#pragma unroll
        for (int j = 0; j < 8; ++j) {
          const int off = kbA + j;
          af.u[j] = lds32A[base + off];
        }
#pragma unroll
        for (int nt = 0; nt < 4; ++nt)
          acc[mi][nt] = __builtin_amdgcn_wmma_f32_16x16x32_bf16(
              false, af.v, false, bf[nt].v, (short)0, acc[mi][nt], false, false);
      }
    }
  }

  // epilogue: D layout: lane -> col n, VGPR j -> row m = kg*8+j -> pixel w
#pragma unroll
  for (int mi = 0; mi < 2; ++mi) {
    const int mt = wv * 2 + mi;
#pragma unroll
    for (int nt = 0; nt < 4; ++nt) {
      const int co = co_base + nt * 16 + n;
      const float bv = bias[co];
      float* yp = Y + (((size_t)b * Cout + co) * HW + mt * 16 + kg * 8);
      float4 o0 = make_float4(acc[mi][nt][0] + bv, acc[mi][nt][1] + bv,
                              acc[mi][nt][2] + bv, acc[mi][nt][3] + bv);
      float4 o1 = make_float4(acc[mi][nt][4] + bv, acc[mi][nt][5] + bv,
                              acc[mi][nt][6] + bv, acc[mi][nt][7] + bv);
      *(float4*)(yp)     = o0;
      *(float4*)(yp + 4) = o1;
    }
  }
}

// ================== weight reorg: [Cout][Cin][5][5] f32 -> [25][Cout][Cin_pad] bf16 ==================
__global__ __launch_bounds__(256) void reorg_w(const float* __restrict__ W,
                                               uint16_t* __restrict__ Wr,
                                               int Cout, int Cin, int Cin_pad, size_t total)
{
  size_t idx = (size_t)blockIdx.x * 256 + threadIdx.x;
  if (idx >= total) return;
  int ci = (int)(idx % Cin_pad);
  size_t t = idx / Cin_pad;
  int co  = (int)(t % Cout);
  int tap = (int)(t / Cout);
  uint16_t v = 0;
  if (ci < Cin) v = f32_to_bf16(W[((size_t)co * Cin + ci) * 25 + tap]);
  Wr[idx] = v;
}

// ================== channel concat (mixed f32/bf16 sources) -> bf16 X ==================
struct ConcatArgs {
  const void* p[5];
  int ch[5];
  int isbf[5];
  int nsrc;
  int Ctot;
  int Cin_pad;
};
__global__ __launch_bounds__(256) void concat_to_bf16(ConcatArgs a, uint16_t* __restrict__ X,
                                                      size_t total)
{
  size_t idx = (size_t)blockIdx.x * 256 + threadIdx.x;
  if (idx >= total) return;
  int pix = (int)(idx & 255);
  size_t t = idx >> 8;
  int ch = (int)(t % a.Cin_pad);
  size_t b = t / a.Cin_pad;
  uint16_t out = 0;
  if (ch < a.Ctot) {
    int s = 0, off = ch;
    while (off >= a.ch[s]) { off -= a.ch[s]; ++s; }
    size_t si = ((size_t)b * a.ch[s] + off) * HW + pix;
    out = a.isbf[s] ? ((const uint16_t*)a.p[s])[si]
                    : f32_to_bf16(((const float*)a.p[s])[si]);
  }
  X[idx] = out;
}

// ================== ConvLSTM gating ==================
__global__ __launch_bounds__(256) void lstm_gate(const float* __restrict__ gates,
                                                 const float* __restrict__ c_prev,
                                                 float* __restrict__ c_out,
                                                 float* __restrict__ h_f32,
                                                 uint16_t* __restrict__ h_bf,
                                                 int C, size_t total)
{
  size_t idx = (size_t)blockIdx.x * 256 + threadIdx.x;
  if (idx >= total) return;
  int pix = (int)(idx & 255);
  size_t t = idx >> 8;
  int ch = (int)(t % C);
  size_t b = t / C;
  size_t gb = (((size_t)b * 4 * C) + ch) * HW + pix;
  const size_t cs = (size_t)C * HW;
  float gi = gates[gb];
  float gf = gates[gb + cs];
  float go = gates[gb + 2 * cs];
  float gg = gates[gb + 3 * cs];
  float c2 = sigm(gf) * c_prev[idx] + sigm(gi) * tanhf(gg);
  float h2 = sigm(go) * tanhf(c2);
  c_out[idx] = c2;
  if (h_f32) h_f32[idx] = h2;
  if (h_bf)  h_bf[idx]  = f32_to_bf16(h2);
}

// ================== reparameterized sample ==================
__global__ __launch_bounds__(256) void sample_z(const float* __restrict__ mlv_q,
                                                const float* __restrict__ eps_t,
                                                float* __restrict__ z,
                                                float* __restrict__ z_out2, size_t total)
{
  size_t idx = (size_t)blockIdx.x * 256 + threadIdx.x;
  if (idx >= total) return;
  int pix = (int)(idx & 255);
  size_t t = idx >> 8;
  int ch = (int)(t % NZ);
  size_t b = t / NZ;
  size_t mi = (((size_t)b * 2 * NZ) + ch) * HW + pix;
  float mq = mlv_q[mi], lq = mlv_q[mi + (size_t)NZ * HW];
  float zv = mq + expf(0.5f * lq) * eps_t[idx];
  z[idx] = zv;
  if (z_out2) z_out2[idx] = zv;
}

// ================== per-batch NLL difference ==================
__global__ __launch_bounds__(256) void lpq_step(const float* __restrict__ mlvq,
                                                const float* __restrict__ mlvp,
                                                const float* __restrict__ z,
                                                float* __restrict__ lpq)
{
  __shared__ float red[256];
  const int b = blockIdx.x;
  float s = 0.0f;
  for (int j = threadIdx.x; j < NZ * HW; j += 256) {
    int ch = j >> 8, pix = j & 255;
    size_t mi = (((size_t)b * 2 * NZ) + ch) * HW + pix;
    float mq = mlvq[mi], lq = mlvq[mi + (size_t)NZ * HW];
    float mp = mlvp[mi], lp = mlvp[mi + (size_t)NZ * HW];
    float zv = z[((size_t)b * NZ + ch) * HW + pix];
    float dq = zv - mq, dp = zv - mp;
    s += 0.5f * ((lq - lp) + dq * dq * expf(-lq) - dp * dp * expf(-lp));
  }
  red[threadIdx.x] = s;
  __syncthreads();
  for (int o = 128; o > 0; o >>= 1) {
    if (threadIdx.x < o) red[threadIdx.x] += red[threadIdx.x + o];
    __syncthreads();
  }
  if (threadIdx.x == 0) lpq[b] += red[0];
}

// ================== KL accumulation (scaled by 0.5/n) ==================
__global__ __launch_bounds__(256) void kl_step(const float* __restrict__ mlvq,
                                               const float* __restrict__ mlvp,
                                               float* __restrict__ kl, size_t total)
{
  __shared__ float red[256];
  float s = 0.0f;
  for (size_t idx = (size_t)blockIdx.x * 256 + threadIdx.x; idx < total;
       idx += (size_t)gridDim.x * 256) {
    int pix = (int)(idx & 255);
    size_t t = idx >> 8;
    int ch = (int)(t % NZ);
    size_t b = t / NZ;
    size_t mi = (((size_t)b * 2 * NZ) + ch) * HW + pix;
    float mq = mlvq[mi], lq = mlvq[mi + (size_t)NZ * HW];
    float mp = mlvp[mi], lp = mlvp[mi + (size_t)NZ * HW];
    float d = mq - mp;
    s += (lp - lq) + (expf(lq) + d * d) * expf(-lp) - 1.0f;
  }
  red[threadIdx.x] = s;
  __syncthreads();
  for (int o = 128; o > 0; o >>= 1) {
    if (threadIdx.x < o) red[threadIdx.x] += red[threadIdx.x + o];
    __syncthreads();
  }
  if (threadIdx.x == 0) atomicAdd(kl, red[0] * (0.5f / (float)BN));
}

// ================== zero fills ==================
__global__ __launch_bounds__(256) void zero_f32(float* p, size_t n) {
  size_t i = (size_t)blockIdx.x * 256 + threadIdx.x;
  if (i < n) p[i] = 0.0f;
}
__global__ __launch_bounds__(256) void zero_u16(uint16_t* p, size_t n) {
  size_t i = (size_t)blockIdx.x * 256 + threadIdx.x;
  if (i < n) p[i] = 0;
}

// ================== host orchestration ==================
static inline int gr(size_t n) { return (int)((n + 255) / 256); }

extern "C" void kernel_launch(void* const* d_in, const int* in_sizes, int n_in,
                              void* d_out, int out_size, void* d_ws, size_t ws_size,
                              hipStream_t stream)
{
  const float* C_t   = (const float*)d_in[0];
  const float* D_t   = (const float*)d_in[1];
  const float* a_tmo = (const float*)d_in[2];
  const float* s_tmo = (const float*)d_in[3];
  const float* c_ssm = (const float*)d_in[4];
  const float* z_tmo = (const float*)d_in[5];
  const float* eps   = (const float*)d_in[6];
  const float* W_ssm = (const float*)d_in[7];   const float* b_ssm = (const float*)d_in[8];
  const float* W_p   = (const float*)d_in[9];   const float* b_p   = (const float*)d_in[10];
  const float* W_q   = (const float*)d_in[11];  const float* b_q   = (const float*)d_in[12];
  const float* W_rp  = (const float*)d_in[13];  const float* b_rp  = (const float*)d_in[14];
  const float* W_rq  = (const float*)d_in[15];  const float* b_rq  = (const float*)d_in[16];

  const int CI_SSM = 328, CP_SSM = 352;
  const int CI_Q   = 520, CP_Q   = 544;
  const int CI_P   = 456, CP_P   = 480;
  const int CI_R   = 128, CP_R   = 128;
  const int CO_BIG = 512, CO_R   = 128;

  // d_out layout: z_t | s_t | c_ssm_t | kl | lpq
  float* o_z   = (float*)d_out;
  float* o_s   = o_z + (size_t)BN * NZ * HW;
  float* o_c   = o_s + (size_t)BN * NCS * HW;
  float* o_kl  = o_c + (size_t)BN * NCS * HW;
  float* o_lpq = o_kl + 1;

  char* w = (char*)d_ws;
  auto carve = [&](size_t bytes) -> void* {
    void* p = (void*)w;
    w += (bytes + 255) & ~(size_t)255;
    return p;
  };
  const size_t nW_ssm = 25ull * CO_BIG * CP_SSM;
  const size_t nW_q   = 25ull * CO_BIG * CP_Q;
  const size_t nW_p   = 25ull * CO_BIG * CP_P;
  const size_t nW_r   = 25ull * CO_R   * CP_R;
  uint16_t* Wr_ssm = (uint16_t*)carve(nW_ssm * 2);
  uint16_t* Wr_q   = (uint16_t*)carve(nW_q * 2);
  uint16_t* Wr_p   = (uint16_t*)carve(nW_p * 2);
  uint16_t* Wr_rp  = (uint16_t*)carve(nW_r * 2);
  uint16_t* Wr_rq  = (uint16_t*)carve(nW_r * 2);

  const size_t nX = (size_t)BN * CP_Q * HW;
  uint16_t* Xbuf  = (uint16_t*)carve(nX * 2);
  const size_t nH = (size_t)BN * NCL * HW;
  uint16_t* hp_bf = (uint16_t*)carve(nH * 2);
  uint16_t* hq_bf = (uint16_t*)carve(nH * 2);

  const size_t nG = (size_t)BN * CO_BIG * HW;
  float* gates = (float*)carve(nG * 4);
  float* cp    = (float*)carve(nH * 4);
  float* cq    = (float*)carve(nH * 4);
  float* mlvp  = (float*)carve(nH * 4);
  float* mlvq  = (float*)carve(nH * 4);
  const size_t nZ = (size_t)BN * NZ * HW;
  float* zbuf  = (float*)carve(nZ * 4);

  reorg_w<<<gr(nW_ssm), 256, 0, stream>>>(W_ssm, Wr_ssm, CO_BIG, CI_SSM, CP_SSM, nW_ssm);
  reorg_w<<<gr(nW_q),   256, 0, stream>>>(W_q,   Wr_q,   CO_BIG, CI_Q,   CP_Q,   nW_q);
  reorg_w<<<gr(nW_p),   256, 0, stream>>>(W_p,   Wr_p,   CO_BIG, CI_P,   CP_P,   nW_p);
  reorg_w<<<gr(nW_r),   256, 0, stream>>>(W_rp,  Wr_rp,  CO_R,   CI_R,   CP_R,   nW_r);
  reorg_w<<<gr(nW_r),   256, 0, stream>>>(W_rq,  Wr_rq,  CO_R,   CI_R,   CP_R,   nW_r);

  zero_f32<<<gr(nH), 256, 0, stream>>>(cp, nH);
  zero_f32<<<gr(nH), 256, 0, stream>>>(cq, nH);
  zero_u16<<<gr(nH), 256, 0, stream>>>(hp_bf, nH);
  zero_u16<<<gr(nH), 256, 0, stream>>>(hq_bf, nH);
  zero_f32<<<1, 256, 0, stream>>>(o_kl, 129);

  // ---- SSM ConvLSTM ----
  {
    ConcatArgs a = {};
    a.p[0] = z_tmo; a.ch[0] = NZ;  a.isbf[0] = 0;
    a.p[1] = C_t;   a.ch[1] = NCC; a.isbf[1] = 0;
    a.p[2] = a_tmo; a.ch[2] = NA;  a.isbf[2] = 0;
    a.p[3] = s_tmo; a.ch[3] = NCS; a.isbf[3] = 0;
    a.nsrc = 4; a.Ctot = CI_SSM; a.Cin_pad = CP_SSM;
    size_t tot = (size_t)BN * CP_SSM * HW;
    concat_to_bf16<<<gr(tot), 256, 0, stream>>>(a, Xbuf, tot);
  }
  conv5x5_wmma<<<dim3(CO_BIG / 64, BN), 256, SMEM_BYTES, stream>>>(Xbuf, Wr_ssm, b_ssm, gates, CP_SSM, CO_BIG);
  lstm_gate<<<gr(nH), 256, 0, stream>>>(gates, c_ssm, o_c, o_s, (uint16_t*)nullptr, NCS, nH);

  // ---- draw-step loop ----
  for (int t = 0; t < STEPS; ++t) {
    const float* eps_t = eps + (size_t)t * nZ;

    {
      ConcatArgs a = {};
      a.p[0] = hp_bf; a.ch[0] = NCL; a.isbf[0] = 1;
      a.p[1] = D_t;   a.ch[1] = NCC; a.isbf[1] = 0;
      a.p[2] = o_s;   a.ch[2] = NCS; a.isbf[2] = 0;
      a.p[3] = a_tmo; a.ch[3] = NA;  a.isbf[3] = 0;
      a.p[4] = hq_bf; a.ch[4] = NCL; a.isbf[4] = 1;
      a.nsrc = 5; a.Ctot = CI_Q; a.Cin_pad = CP_Q;
      size_t tot = (size_t)BN * CP_Q * HW;
      concat_to_bf16<<<gr(tot), 256, 0, stream>>>(a, Xbuf, tot);
    }
    conv5x5_wmma<<<dim3(CO_BIG / 64, BN), 256, SMEM_BYTES, stream>>>(Xbuf, Wr_q, b_q, gates, CP_Q, CO_BIG);
    lstm_gate<<<gr(nH), 256, 0, stream>>>(gates, cq, cq, (float*)nullptr, hq_bf, NCL, nH);

    conv5x5_wmma<<<dim3(CO_R / 64, BN), 256, SMEM_BYTES, stream>>>(hq_bf, Wr_rq, b_rq, mlvq, CP_R, CO_R);
    conv5x5_wmma<<<dim3(CO_R / 64, BN), 256, SMEM_BYTES, stream>>>(hp_bf, Wr_rp, b_rp, mlvp, CP_R, CO_R);

    sample_z<<<gr(nZ), 256, 0, stream>>>(mlvq, eps_t, zbuf,
                                         (t == STEPS - 1) ? o_z : (float*)nullptr, nZ);

    lpq_step<<<BN, 256, 0, stream>>>(mlvq, mlvp, zbuf, o_lpq);
    kl_step<<<256, 256, 0, stream>>>(mlvq, mlvp, o_kl, nZ);

    {
      ConcatArgs a = {};
      a.p[0] = zbuf;  a.ch[0] = NZ;  a.isbf[0] = 0;
      a.p[1] = C_t;   a.ch[1] = NCC; a.isbf[1] = 0;
      a.p[2] = o_s;   a.ch[2] = NCS; a.isbf[2] = 0;
      a.p[3] = a_tmo; a.ch[3] = NA;  a.isbf[3] = 0;
      a.p[4] = hp_bf; a.ch[4] = NCL; a.isbf[4] = 1;
      a.nsrc = 5; a.Ctot = CI_P; a.Cin_pad = CP_P;
      size_t tot = (size_t)BN * CP_P * HW;
      concat_to_bf16<<<gr(tot), 256, 0, stream>>>(a, Xbuf, tot);
    }
    conv5x5_wmma<<<dim3(CO_BIG / 64, BN), 256, SMEM_BYTES, stream>>>(Xbuf, Wr_p, b_p, gates, CP_P, CO_BIG);
    lstm_gate<<<gr(nH), 256, 0, stream>>>(gates, cp, cp, (float*)nullptr, hp_bf, NCL, nH);
  }

  (void)in_sizes; (void)n_in; (void)out_size; (void)ws_size;
}
